// LSTMCRFTagger_11493332484361
// MI455X (gfx1250) — matrix operand order repaired and checked by reading, not compiled, not run on hardware
//
#include <hip/hip_runtime.h>
#include <hip/hip_bf16.h>

#define B_  64
#define L_  128
#define E_  300
#define EP_ 320          // E padded to multiple of 32 for WMMA K
#define H_  256
#define G4_ 1024         // 4*H gate width
#define HD_ 512
#define T_  64
#define M_  (B_*L_)      // 8192 flattened (batch, pos) rows

// lstm_rec LDS layout (bytes) — host uses the constant total unconditionally
#define GATES_BYTES (16 * G4_ * 4)        // 65536
#define HBF_OFF     GATES_BYTES
#define HBF_BYTES   (16 * H_ * 2)         // 8192
#define XP_OFF      (HBF_OFF + HBF_BYTES) // 73728
#define XP_BYTES    (2 * 16 * G4_ * 4)    // 131072 (double buffer)
#define LSTM_LDS    (XP_OFF + XP_BYTES)   // 204800 < 320KB

typedef __bf16 bf16x8  __attribute__((ext_vector_type(8)));
typedef __bf16 bf16x16 __attribute__((ext_vector_type(16)));
typedef float  f32x8   __attribute__((ext_vector_type(8)));
typedef int    v4i     __attribute__((ext_vector_type(4)));

#if defined(__AMDGCN__) && __has_builtin(__builtin_amdgcn_global_load_async_to_lds_b128)
#define HAVE_ASYNC_LDS 1
#endif

static __device__ inline __bf16 to_bf16(float f) {
  unsigned u; __builtin_memcpy(&u, &f, 4);
  unsigned r = u + 0x7FFFu + ((u >> 16) & 1u);          // round-to-nearest-even
  unsigned short s = (unsigned short)(r >> 16);
  __bf16 b; __builtin_memcpy(&b, &s, 2); return b;
}
static __device__ inline float sigf(float x) { return 1.0f / (1.0f + expf(-x)); }

static __device__ inline f32x8 wmma_bf16(bf16x16 a, bf16x16 b, f32x8 c) {
  // emits v_wmma_f32_16x16x32_bf16
  return __builtin_amdgcn_wmma_f32_16x16x32_bf16(false, a, false, b, (short)0, c, false, false);
}

#ifdef HAVE_ASYNC_LDS
static __device__ inline void async_cp16(const float* gsrc, float* ldst) {
  __builtin_amdgcn_global_load_async_to_lds_b128(
      (__attribute__((address_space(1))) v4i*)gsrc,
      (__attribute__((address_space(3))) v4i*)ldst, 0, 0);
}
static __device__ inline void wait_async0() {
#if __has_builtin(__builtin_amdgcn_s_wait_asynccnt)
  __builtin_amdgcn_s_wait_asynccnt(0);
#else
  asm volatile("s_wait_asynccnt 0x0" ::: "memory");
#endif
}
#endif

// A tile (16x32, 16-bit): lanes 0-15 hold M=lane, lanes 16-31 hold M=lane-16.
// VGPR 0-3: K = kb..kb+7 ; VGPR 4-7: K = kb+8..kb+15 with kb = (lane<16)?0:16.
static __device__ inline bf16x16 load_a(const __bf16* src, int ld, int r0, int k0, int lane) {
  int row = r0 + (lane & 15);
  int kb  = (lane < 16) ? 0 : 16;
  bf16x8 lo = *(const bf16x8*)(src + (size_t)row * ld + k0 + kb);
  bf16x8 hi = *(const bf16x8*)(src + (size_t)row * ld + k0 + 8 + kb);
  bf16x16 a;
  #pragma unroll
  for (int i = 0; i < 8; ++i) { a[i] = lo[i]; a[i + 8] = hi[i]; }
  return a;
}
// B tile (32x16, 16-bit) from transposed operand stored [N][K] row-major:
// lane gives N = n0 + lane%16; lanes 0-15 hold K = k0..k0+15, lanes 16-31 K = k0+16..k0+31.
static __device__ inline bf16x16 load_bt(const __bf16* src, int ld, int n0, int k0, int lane) {
  int n    = n0 + (lane & 15);
  int koff = (lane < 16) ? 0 : 16;
  return *(const bf16x16*)(src + (size_t)n * ld + k0 + koff);
}

// ---------------- prep kernels ----------------
__global__ void conv_wih(const float* __restrict__ wf, const float* __restrict__ wb,
                         __bf16* __restrict__ dst) {
  int idx = blockIdx.x * blockDim.x + threadIdx.x;
  if (idx >= 2 * G4_ * EP_) return;
  int d = idx / (G4_ * EP_);
  int rem = idx % (G4_ * EP_);
  int g = rem / EP_, k = rem % EP_;
  const float* src = d ? wb : wf;
  float v = (k < E_) ? src[(size_t)g * E_ + k] : 0.0f;
  dst[idx] = to_bf16(v);
}

__global__ void conv2_bf16(const float* __restrict__ a, const float* __restrict__ b,
                           __bf16* __restrict__ dst, int per) {
  int idx = blockIdx.x * blockDim.x + threadIdx.x;
  if (idx >= 2 * per) return;
  const float* s = (idx < per) ? a : b;
  int k = (idx < per) ? idx : idx - per;
  dst[idx] = to_bf16(s[k]);
}

__global__ void conv1_bf16(const float* __restrict__ a, __bf16* __restrict__ dst, int n) {
  int idx = blockIdx.x * blockDim.x + threadIdx.x;
  if (idx < n) dst[idx] = to_bf16(a[idx]);
}

__global__ void gather_x(const int* __restrict__ sent, const float* __restrict__ emb,
                         __bf16* __restrict__ xb) {
  int idx = blockIdx.x * blockDim.x + threadIdx.x;
  if (idx >= M_ * EP_) return;
  int m = idx / EP_, k = idx % EP_;
  float v = (k < E_) ? emb[(size_t)sent[m] * E_ + k] : 0.0f;
  xb[idx] = to_bf16(v);
}

// ---------------- input projection GEMM: xp[m][g] = x[m][:]·wih[g][:] + bih[g]+bhh[g] ----------------
// Each wave computes TWO adjacent 16x16 N-tiles, reusing the A fragment.
__global__ void xproj_gemm(const __bf16* __restrict__ xb, const __bf16* __restrict__ wih,
                           const float* __restrict__ bih, const float* __restrict__ bhh,
                           float* __restrict__ xp) {
  int wave = threadIdx.x >> 5, lane = threadIdx.x & 31;
  int mt = blockIdx.x;                         // 512 M-tiles
  int nt0 = (blockIdx.y * 8 + wave) * 2;       // 64 N-tiles, 2 per wave
  f32x8 acc0 = {0.f,0.f,0.f,0.f,0.f,0.f,0.f,0.f};
  f32x8 acc1 = {0.f,0.f,0.f,0.f,0.f,0.f,0.f,0.f};
  for (int k0 = 0; k0 < EP_; k0 += 32) {
    bf16x16 a  = load_a(xb, EP_, mt * 16, k0, lane);
    bf16x16 b0 = load_bt(wih, EP_, nt0 * 16, k0, lane);
    bf16x16 b1 = load_bt(wih, EP_, (nt0 + 1) * 16, k0, lane);
    if (k0 + 64 < EP_) {   // global_prefetch_b8 hint for the B stream
      __builtin_prefetch((const void*)(wih + (size_t)(nt0 * 16 + (lane & 15)) * EP_ + k0 + 64), 0, 0);
    }
    acc0 = wmma_bf16(a, b0, acc0);
    acc1 = wmma_bf16(a, b1, acc1);
  }
  int g0 = nt0 * 16 + (lane & 15);
  int g1 = g0 + 16;
  float bias0 = bih[g0] + bhh[g0];
  float bias1 = bih[g1] + bhh[g1];
  int mbase = mt * 16 + ((lane >= 16) ? 8 : 0);
  #pragma unroll
  for (int r = 0; r < 8; ++r) {
    xp[(size_t)(mbase + r) * G4_ + g0] = acc0[r] + bias0;
    xp[(size_t)(mbase + r) * G4_ + g1] = acc1[r] + bias1;
  }
}

// ---------------- persistent LSTM recurrence ----------------
// grid (2 dirs, 4 batch tiles) x 256 threads. Each block owns 16 batch rows.
// Per step: 64 WMMA N-tiles over 8 waves; xp slice for step t+1 is double-buffered
// into LDS with GLOBAL_LOAD_ASYNC_TO_LDS_B128 overlapped with the WMMA phase.
__global__ void __launch_bounds__(256, 1)
lstm_rec(const float* __restrict__ xp_f, const float* __restrict__ xp_b,
         const __bf16* __restrict__ whh_f, const __bf16* __restrict__ whh_b,
         __bf16* __restrict__ hs) {
  extern __shared__ char smem[];
  float*  gates = (float*)smem;                 // [16][1024]
  __bf16* hbf   = (__bf16*)(smem + HBF_OFF);    // [16][256]
  float*  xpb   = (float*)(smem + XP_OFF);      // [2][16][1024]
  int d   = blockIdx.x;
  int mtg = blockIdx.y;                          // batch tile: rows mtg*16..mtg*16+15
  const float*  xp  = d == 0 ? xp_f : xp_b;
  const __bf16* whh = d == 0 ? whh_f : whh_b;
  int tid = threadIdx.x, wave = tid >> 5, lane = tid & 31;

  for (int i = tid; i < 16 * H_; i += 256) hbf[i] = to_bf16(0.0f);
  float c[16];
  #pragma unroll
  for (int i = 0; i < 16; ++i) c[i] = 0.0f;
  int lrB = tid >> 4;             // local batch row owned in phase B
  int jb  = (tid & 15) << 4;      // 16 hidden units owned

#ifdef HAVE_ASYNC_LDS
  { // preload xp slice for the first step into buffer 0
    int t0 = (d == 0) ? 0 : (L_ - 1);
    for (int q = tid; q < 4096; q += 256) {
      int r = q >> 8, col = (q & 255) << 2;
      async_cp16(xp + (((size_t)(mtg * 16 + r)) * L_ + t0) * G4_ + col,
                 xpb + r * G4_ + col);
    }
    wait_async0();
  }
#endif
  __syncthreads();

  for (int s = 0; s < L_; ++s) {
    int t = (d == 0) ? s : (L_ - 1 - s);
#ifdef HAVE_ASYNC_LDS
    if (s + 1 < L_) {  // overlap next step's xp copy with this step's WMMAs
      int tn = (d == 0) ? (s + 1) : (L_ - 2 - s);
      float* dstb = xpb + ((s + 1) & 1) * (16 * G4_);
      for (int q = tid; q < 4096; q += 256) {
        int r = q >> 8, col = (q & 255) << 2;
        async_cp16(xp + (((size_t)(mtg * 16 + r)) * L_ + tn) * G4_ + col,
                   dstb + r * G4_ + col);
      }
    }
    const float* xcur = xpb + (s & 1) * (16 * G4_);
#endif
    // phase A: gates = h @ whh^T + xp[:,t,:]  — 64 WMMA tiles over 8 waves
    for (int i = 0; i < 8; ++i) {
      int nt = wave * 8 + i;
      f32x8 acc = {0.f,0.f,0.f,0.f,0.f,0.f,0.f,0.f};
      for (int k0 = 0; k0 < H_; k0 += 32) {
        bf16x16 a = load_a(hbf, H_, 0, k0, lane);           // LDS-resident h
        bf16x16 b = load_bt(whh, H_, nt * 16, k0, lane);    // weights from L2
        acc = wmma_bf16(a, b, acc);
      }
      int g  = nt * 16 + (lane & 15);
      int mb = (lane >= 16) ? 8 : 0;
      #pragma unroll
      for (int r = 0; r < 8; ++r) {
        int lr = mb + r;    // local batch row
#ifdef HAVE_ASYNC_LDS
        gates[lr * G4_ + g] = acc[r] + xcur[lr * G4_ + g];
#else
        gates[lr * G4_ + g] = acc[r] +
            xp[(((size_t)(mtg * 16 + lr)) * L_ + t) * G4_ + g];
#endif
      }
    }
    __syncthreads();
    // phase B: elementwise LSTM cell update, c in registers
    #pragma unroll
    for (int jj = 0; jj < 16; ++jj) {
      int j = jb + jj;
      float gi = gates[lrB * G4_ + j];
      float gf = gates[lrB * G4_ + j + H_];
      float gg = gates[lrB * G4_ + j + 2 * H_];
      float go = gates[lrB * G4_ + j + 3 * H_];
      float cn = sigf(gf) * c[jj] + sigf(gi) * tanhf(gg);
      c[jj] = cn;
      float h = sigf(go) * tanhf(cn);
      __bf16 hv = to_bf16(h);
      hbf[lrB * H_ + j] = hv;
      hs[(((size_t)(mtg * 16 + lrB)) * L_ + t) * HD_ + d * H_ + j] = hv;
    }
#ifdef HAVE_ASYNC_LDS
    wait_async0();          // next-step xp buffer complete before barrier
#endif
    __syncthreads();
  }
}

// ---------------- output projection: em[m][n] = hs[m][:]·wout[n][:] + bout[n] ----------------
__global__ void emis_gemm(const __bf16* __restrict__ hs, const __bf16* __restrict__ wout,
                          const float* __restrict__ bout, float* __restrict__ em) {
  int wave = threadIdx.x >> 5, lane = threadIdx.x & 31;
  int mt = blockIdx.x * 2 + (wave >> 2);   // 512 M-tiles, 2 per block
  int nt = wave & 3;                        // 4 N-tiles
  f32x8 acc = {0.f,0.f,0.f,0.f,0.f,0.f,0.f,0.f};
  for (int k0 = 0; k0 < HD_; k0 += 32) {
    bf16x16 a = load_a(hs, HD_, mt * 16, k0, lane);
    bf16x16 b = load_bt(wout, HD_, nt * 16, k0, lane);
    acc = wmma_bf16(a, b, acc);
  }
  int n = nt * 16 + (lane & 15);
  float bias = bout[n];
  int mb = mt * 16 + ((lane >= 16) ? 8 : 0);
  #pragma unroll
  for (int r = 0; r < 8; ++r)
    em[(size_t)(mb + r) * T_ + n] = acc[r] + bias;
}

// ---------------- CRF forward algorithm + numerator, one block per batch ----------------
__global__ void crf_nll(const float* __restrict__ em, const int* __restrict__ tags,
                        const float* __restrict__ start_t, const float* __restrict__ end_t,
                        const float* __restrict__ trans, float* __restrict__ nll) {
  __shared__ float score[T_];
  int b = blockIdx.x, j = threadIdx.x;
  const float* emb_ = em + (size_t)b * L_ * T_;
  score[j] = start_t[j] + emb_[j];
  __syncthreads();
  for (int t = 1; t < L_; ++t) {
    float m = -3.4e38f;
    #pragma unroll 4
    for (int i = 0; i < T_; ++i) m = fmaxf(m, score[i] + trans[i * T_ + j]);
    float s = 0.f;
    #pragma unroll 4
    for (int i = 0; i < T_; ++i) s += expf(score[i] + trans[i * T_ + j] - m);
    float nxt = m + logf(s) + emb_[t * T_ + j];
    __syncthreads();
    score[j] = nxt;
    __syncthreads();
  }
  if (j == 0) {
    float m = -3.4e38f;
    for (int i = 0; i < T_; ++i) m = fmaxf(m, score[i] + end_t[i]);
    float s = 0.f;
    for (int i = 0; i < T_; ++i) s += expf(score[i] + end_t[i] - m);
    float logz = m + logf(s);
    const int* tg = tags + b * L_;
    float num = start_t[tg[0]] + end_t[tg[L_ - 1]];
    for (int t = 0; t < L_; ++t)     num += emb_[t * T_ + tg[t]];
    for (int t = 0; t < L_ - 1; ++t) num += trans[tg[t] * T_ + tg[t + 1]];
    nll[b] = logz - num;
  }
}

__global__ void reduce_mean(const float* __restrict__ nll, float* __restrict__ out) {
  if (threadIdx.x == 0) {
    float s = 0.f;
    for (int i = 0; i < B_; ++i) s += nll[i];
    out[0] = s * (1.0f / B_);
  }
}

extern "C" void kernel_launch(void* const* d_in, const int* in_sizes, int n_in,
                              void* d_out, int out_size, void* d_ws, size_t ws_size,
                              hipStream_t stream) {
  (void)in_sizes; (void)n_in; (void)out_size; (void)ws_size;
  const int*   sent   = (const int*)d_in[0];
  const int*   tags   = (const int*)d_in[1];
  const float* emb    = (const float*)d_in[2];
  const float* wihf   = (const float*)d_in[3];
  const float* whhf   = (const float*)d_in[4];
  const float* bihf   = (const float*)d_in[5];
  const float* bhhf   = (const float*)d_in[6];
  const float* wihb   = (const float*)d_in[7];
  const float* whhb   = (const float*)d_in[8];
  const float* bihb   = (const float*)d_in[9];
  const float* bhhb   = (const float*)d_in[10];
  const float* wout   = (const float*)d_in[11];
  const float* bout   = (const float*)d_in[12];
  const float* startt = (const float*)d_in[13];
  const float* endt   = (const float*)d_in[14];
  const float* trans  = (const float*)d_in[15];
  float* out = (float*)d_out;

  char* ws = (char*)d_ws;
  size_t off = 0;
  auto take = [&](size_t bytes) { char* p = ws + off; off = (off + bytes + 255) & ~(size_t)255; return p; };
  __bf16* xb    = (__bf16*)take((size_t)M_ * EP_ * 2);
  __bf16* wihB  = (__bf16*)take((size_t)2 * G4_ * EP_ * 2);
  __bf16* whhB  = (__bf16*)take((size_t)2 * G4_ * H_ * 2);
  __bf16* woutB = (__bf16*)take((size_t)T_ * HD_ * 2);
  float*  xp    = (float*)take((size_t)2 * M_ * G4_ * 4);
  __bf16* hs    = (__bf16*)take((size_t)M_ * HD_ * 2);
  float*  em    = (float*)take((size_t)M_ * T_ * 4);
  float*  nll   = (float*)take((size_t)B_ * 4);

  { int n = 2 * G4_ * EP_; conv_wih <<<(n + 255) / 256, 256, 0, stream>>>(wihf, wihb, wihB); }
  { int n = 2 * G4_ * H_;  conv2_bf16<<<(n + 255) / 256, 256, 0, stream>>>(whhf, whhb, whhB, G4_ * H_); }
  { int n = T_ * HD_;      conv1_bf16<<<(n + 255) / 256, 256, 0, stream>>>(wout, woutB, n); }
  { int n = M_ * EP_;      gather_x  <<<(n + 255) / 256, 256, 0, stream>>>(sent, emb, xb); }

  dim3 g3(M_ / 16, 4);   // 512 M-tiles x 4; 8 waves/block, 2 N-tiles/wave
  xproj_gemm<<<g3, 256, 0, stream>>>(xb, wihB,                       bihf, bhhf, xp);
  xproj_gemm<<<g3, 256, 0, stream>>>(xb, wihB + (size_t)G4_ * EP_,   bihb, bhhb, xp + (size_t)M_ * G4_);

  lstm_rec<<<dim3(2, 4), 256, LSTM_LDS, stream>>>(xp, xp + (size_t)M_ * G4_,
                                                  whhB, whhB + (size_t)G4_ * H_, hs);

  emis_gemm<<<(M_ / 16) / 2, 256, 0, stream>>>(hs, woutB, bout, em);

  crf_nll<<<B_, T_, 0, stream>>>(em, tags, startt, endt, trans, nll);
  reduce_mean<<<1, 32, 0, stream>>>(nll, out);
}
